// ClassicAttention_37349035606802
// MI455X (gfx1250) — matrix operand order repaired and checked by reading, not compiled, not run on hardware
//
#include <hip/hip_runtime.h>
#include <hip/hip_bf16.h>

// ---- problem constants -----------------------------------------------------
#define B_     1024
#define NTOK   68
#define C_     768
#define H_     12
#define HD     64
#define N3     2304      // 3*C
#define TOKPAD 96        // token padding for V^T (3 k-steps of 32)
#define MTOT   (B_ * NTOK)   // 69632 = 544 * 128, exact

typedef __attribute__((ext_vector_type(16))) _Float16 v16h;
typedef __attribute__((ext_vector_type(8)))  float    v8f;

struct V16U { union { v16h v; float4 f[2]; }; };

static __device__ __forceinline__ v8f wmma_f16(v16h a, v16h b, v8f c) {
  // D = A(16x32 f16) * B(32x16 f16) + C(16x16 f32)
  return __builtin_amdgcn_wmma_f32_16x16x32_f16(false, a, false, b, (short)0, c,
                                                false, false);
}

// A operand: row-major (M x K), 16x32 tile.
// lane L: row = L&15 (clamped), chunks at k = hi*8 and k = 16 + hi*8.
static __device__ __forceinline__ v16h load_a_rowmajor(const _Float16* base,
                                                       int stride, int lane,
                                                       int rowmax) {
  int row = lane & 15; if (row > rowmax) row = rowmax;
  int hi  = lane >> 4;
  const _Float16* p = base + (long)row * stride + hi * 8;
  V16U u;
  u.f[0] = *(const float4*)(p);       // K 0..7   (+8 for hi half-wave)
  u.f[1] = *(const float4*)(p + 16);  // K 16..23 (+8 for hi half-wave)
  return u.v;
}

// B operand: storage is (N x K) row-major (i.e. weight (out,in)), 32x16 tile.
// lane L: N-row = L&15 (clamped), chunks at k = hi*8 and k = 16 + hi*8.
static __device__ __forceinline__ v16h load_b_nk(const _Float16* base,
                                                 int stride, int lane,
                                                 int rowmax) {
  int n  = lane & 15; if (n > rowmax) n = rowmax;
  int hi = lane >> 4;
  const _Float16* p = base + (long)n * stride + hi * 8;
  V16U u;
  u.f[0] = *(const float4*)(p);
  u.f[1] = *(const float4*)(p + 16);
  return u.v;
}

// ---- kernel 1: x += pe (center-cropped), convert to f16 --------------------
__global__ void prep_x_kernel(const float* __restrict__ x,
                              const float* __restrict__ pe,
                              _Float16* __restrict__ xh) {
  long i = ((long)blockIdx.x * blockDim.x + threadIdx.x);
  long total = (long)MTOT * C_ / 4;
  if (i >= total) return;
  long e = i * 4;
  int  c   = (int)(e % C_);
  long t   = e / C_;
  int  tok = (int)(t % NTOK);
  float4 xv = *(const float4*)(x + e);
  if (tok >= 4) {                       // strt = 128/2 - 64/2 = 32
    const float* pr = pe + (long)(32 + tok - 4) * C_ + c;
    xv.x += pr[0]; xv.y += pr[1]; xv.z += pr[2]; xv.w += pr[3];
  }
  union { _Float16 h[4]; float2 f; } o;
  o.h[0] = (_Float16)xv.x; o.h[1] = (_Float16)xv.y;
  o.h[2] = (_Float16)xv.z; o.h[3] = (_Float16)xv.w;
  *(float2*)(xh + e) = o.f;
}

// ---- kernel 2: convert both weight matrices to f16 -------------------------
__global__ void prep_w_kernel(const float* __restrict__ wq,
                              const float* __restrict__ wp,
                              _Float16* __restrict__ wqh,
                              _Float16* __restrict__ wph) {
  long i  = ((long)blockIdx.x * blockDim.x + threadIdx.x) * 4;
  long nq = (long)N3 * C_;
  long tot = nq + (long)C_ * C_;
  if (i >= tot) return;
  const float* src; _Float16* dst; long e;
  if (i < nq) { src = wq; dst = wqh; e = i; }
  else        { src = wp; dst = wph; e = i - nq; }
  float4 v = *(const float4*)(src + e);
  union { _Float16 h[4]; float2 f; } o;
  o.h[0] = (_Float16)v.x; o.h[1] = (_Float16)v.y;
  o.h[2] = (_Float16)v.z; o.h[3] = (_Float16)v.w;
  *(float2*)(dst + e) = o.f;
}

// ---- kernel 2b: zero V^T token-padding columns 68..95 ----------------------
__global__ void vt_pad_kernel(_Float16* __restrict__ vt) {
  // rows = B*H*HD = 786432, 14 dword-pairs per row (cols 68..95)
  long i = (long)blockIdx.x * blockDim.x + threadIdx.x;
  long total = (long)B_ * H_ * HD * 14;
  if (i >= total) return;
  long row = i / 14;
  int  c   = (int)(i % 14) * 2 + NTOK;
  *(unsigned int*)(vt + row * TOKPAD + c) = 0u;
}

// ---- GEMM core: 32(M) x 64(N) per wave, K=768, two named buffers -----------
// Buffers alternate roles per 64-wide double-step: no register rotation moves,
// no WMMA->VALU WAR hazard nops; loads stay a full compute-phase ahead.
#define GEMM_LOAD(A0, A1, BT, KOFF)                                            \
  do {                                                                         \
    A0 = load_a_rowmajor(arow + (KOFF),                 C_, lane, 15);         \
    A1 = load_a_rowmajor(arow + (long)16 * C_ + (KOFF), C_, lane, 15);         \
    _Pragma("unroll")                                                          \
    for (int j = 0; j < 4; ++j)                                                \
      BT[j] = load_b_nk(brow + (long)(j * 16) * C_ + (KOFF), C_, lane, 15);    \
  } while (0)

#define GEMM_COMPUTE(A0, A1, BT)                                               \
  do {                                                                         \
    _Pragma("unroll")                                                          \
    for (int j = 0; j < 4; ++j) acc[j]     = wmma_f16(A0, BT[j], acc[j]);      \
    _Pragma("unroll")                                                          \
    for (int j = 0; j < 4; ++j) acc[4 + j] = wmma_f16(A1, BT[j], acc[4 + j]);  \
  } while (0)

// ---- kernel 3: QKV GEMM ----------------------------------------------------
// D(69632 x 2304) = xh @ w_qkv^T + b.
// grid (544 m-blocks of 128 rows, 36 n-groups), block 128: 4 waves share the
// same 64-column weight slab (WGP$ reuse), each takes a 32-row slice.
// Writes q,k as (B,H,68,64) f16 (q pre-scaled by hd^-0.5) and V^T as
// (B,H,64,TOKPAD) f16 (real tokens only; padding pre-zeroed).
__global__ __launch_bounds__(128)
void qkv_gemm_kernel(const _Float16* __restrict__ xh,
                     const _Float16* __restrict__ wq,
                     const float* __restrict__ bias,
                     _Float16* __restrict__ qb,
                     _Float16* __restrict__ kb,
                     _Float16* __restrict__ vt) {
  int lane = threadIdx.x & 31;
  int wave = threadIdx.x >> 5;
  int t0   = blockIdx.x * 128 + wave * 32;      // global row of this wave
  int nbase = blockIdx.y * 64;
  int li = lane & 15, hi = lane >> 4;

  const _Float16* arow = xh + (long)t0 * C_;
  const _Float16* brow = wq + (long)nbase * C_;

  v8f acc[8] = {};
  v16h a0x, a1x, a0y, a1y, btx[4], bty[4];

  GEMM_LOAD(a0x, a1x, btx, 0);                  // k = 0 into buffer X
#pragma unroll 1
  for (int kk = 0; kk + 64 < C_; kk += 64) {    // 11 steady-state double-steps
    GEMM_LOAD(a0y, a1y, bty, kk + 32);          // issue k+32 into Y
    __builtin_prefetch(brow + kk + 64, 0, 1);
    GEMM_COMPUTE(a0x, a1x, btx);                // consume X (k)
    GEMM_LOAD(a0x, a1x, btx, kk + 64);          // issue k+64 into X
    GEMM_COMPUTE(a0y, a1y, bty);                // consume Y (k+32)
  }
  GEMM_LOAD(a0y, a1y, bty, C_ - 32);            // k = 736 into Y
  GEMM_COMPUTE(a0x, a1x, btx);                  // k = 704
  GEMM_COMPUTE(a0y, a1y, bty);                  // k = 736

#pragma unroll
  for (int ms = 0; ms < 2; ++ms) {
#pragma unroll
    for (int j = 0; j < 4; ++j) {
      int col = nbase + j * 16;
      int sel = col / C_, rem = col % C_;
      int h = rem / HD, d = rem % HD + li;
      float bv = bias[col + li];
      float scale = (sel == 0) ? 0.125f : 1.0f;  // hd^-0.5 = 64^-0.5
#pragma unroll
      for (int r = 0; r < 8; ++r) {
        int t = t0 + ms * 16 + r + 8 * hi;
        int b = t / NTOK, tok = t % NTOK;        // mul-shift, no real div
        float v = acc[ms * 4 + j][r] + bv;
        if (sel == 2) {
          vt[(((long)(b * H_ + h)) * HD + d) * TOKPAD + tok] = (_Float16)v;
        } else {
          _Float16* dst = (sel == 0 ? qb : kb);
          dst[(((long)(b * H_ + h)) * NTOK + tok) * HD + d] =
              (_Float16)(v * scale);
        }
      }
    }
  }
}

// ---- kernel 4: fused attention, one wave per (b,h) -------------------------
__global__ __launch_bounds__(32)
void attn_kernel(const _Float16* __restrict__ qb,
                 const _Float16* __restrict__ kb,
                 const _Float16* __restrict__ vt,
                 const float* __restrict__ mask,
                 _Float16* __restrict__ aout) {
  __shared__ __attribute__((aligned(16))) float    sS[80 * 80];
  __shared__ __attribute__((aligned(16))) _Float16 sP[80 * TOKPAD];

  int bh = blockIdx.x;
  int b = bh / H_, h = bh % H_;
  int lane = threadIdx.x;
  int li = lane & 15, hi = lane >> 4;

  const _Float16* qh = qb + ((long)bh * NTOK) * HD;
  const _Float16* kh = kb + ((long)bh * NTOK) * HD;
  const _Float16* vh = vt + ((long)bh * HD) * TOKPAD;

  // ---- S = Q * K^T  (25 tiles, 2 k-steps each) ----
  for (int mt = 0; mt < 5; ++mt) {
    int m0 = mt * 16, rmax = 67 - m0;
    v16h a0 = load_a_rowmajor(qh + (long)m0 * HD + 0,  HD, lane, rmax);
    v16h a1 = load_a_rowmajor(qh + (long)m0 * HD + 32, HD, lane, rmax);
#pragma unroll
    for (int nt = 0; nt < 5; ++nt) {
      int n0 = nt * 16, nmax = 67 - n0;
      v16h b0 = load_b_nk(kh + (long)n0 * HD + 0,  HD, lane, nmax);
      v16h b1 = load_b_nk(kh + (long)n0 * HD + 32, HD, lane, nmax);
      v8f c = {};
      c = wmma_f16(a0, b0, c);
      c = wmma_f16(a1, b1, c);
#pragma unroll
      for (int r = 0; r < 8; ++r)
        sS[(m0 + r + 8 * hi) * 80 + n0 + li] = c[r];
    }
  }
  __syncthreads();

  // ---- softmax rows (+ additive mask, per reference) ----
  const float* mbase = mask + (long)b * NTOK * NTOK;
  for (int row = lane; row < 80; row += 32) {
    if (row < NTOK) {
      const float* mr = mbase + (long)row * NTOK;
      float mx = -3.0e38f;
      for (int n = 0; n < NTOK; ++n) {
        float v = sS[row * 80 + n] + mr[n];
        sS[row * 80 + n] = v;
        mx = v > mx ? v : mx;
      }
      float sum = 0.f;
      for (int n = 0; n < NTOK; ++n) {
        float e = __expf(sS[row * 80 + n] - mx);
        sum += e;
        sS[row * 80 + n] = e;
      }
      float inv = 1.0f / sum;
      for (int n = 0; n < NTOK; ++n)
        sP[row * TOKPAD + n] = (_Float16)(sS[row * 80 + n] * inv);
      for (int n = NTOK; n < TOKPAD; ++n) sP[row * TOKPAD + n] = (_Float16)0.f;
    } else {
      for (int n = 0; n < TOKPAD; ++n) sP[row * TOKPAD + n] = (_Float16)0.f;
    }
  }
  __syncthreads();

  // ---- O = P * V  (P from LDS, V^T from global; 3 k-steps over 96 toks) ----
  for (int mt = 0; mt < 5; ++mt) {
    int m0 = mt * 16;
    v16h pa[3];
#pragma unroll
    for (int kk = 0; kk < 3; ++kk)
      pa[kk] = load_a_rowmajor(&sP[m0 * TOKPAD + kk * 32], TOKPAD, lane, 15);
#pragma unroll
    for (int nt = 0; nt < 4; ++nt) {
      v8f c = {};
#pragma unroll
      for (int kk = 0; kk < 3; ++kk) {
        v16h bt = load_b_nk(vh + (long)(nt * 16) * TOKPAD + kk * 32,
                            TOKPAD, lane, 15);
        c = wmma_f16(pa[kk], bt, c);
      }
      int d = h * HD + nt * 16 + li;
#pragma unroll
      for (int r = 0; r < 8; ++r) {
        int tok = m0 + r + 8 * hi;
        if (tok < NTOK)
          aout[((long)(b * NTOK + tok)) * C_ + d] = (_Float16)c[r];
      }
    }
  }
}

// ---- kernel 5: output projection + bias (fp32 out) -------------------------
// Same 32x64-per-wave double-buffered core; output rows are flat (MTOT,768).
__global__ __launch_bounds__(128)
void proj_gemm_kernel(const _Float16* __restrict__ ain,
                      const _Float16* __restrict__ wp,
                      const float* __restrict__ bias,
                      float* __restrict__ out) {
  int lane = threadIdx.x & 31;
  int wave = threadIdx.x >> 5;
  int t0   = blockIdx.x * 128 + wave * 32;
  int nbase = blockIdx.y * 64;
  int li = lane & 15, hi = lane >> 4;

  const _Float16* arow = ain + (long)t0 * C_;
  const _Float16* brow = wp + (long)nbase * C_;

  v8f acc[8] = {};
  v16h a0x, a1x, a0y, a1y, btx[4], bty[4];

  GEMM_LOAD(a0x, a1x, btx, 0);
#pragma unroll 1
  for (int kk = 0; kk + 64 < C_; kk += 64) {
    GEMM_LOAD(a0y, a1y, bty, kk + 32);
    __builtin_prefetch(brow + kk + 64, 0, 1);
    GEMM_COMPUTE(a0x, a1x, btx);
    GEMM_LOAD(a0x, a1x, btx, kk + 64);
    GEMM_COMPUTE(a0y, a1y, bty);
  }
  GEMM_LOAD(a0y, a1y, bty, C_ - 32);
  GEMM_COMPUTE(a0x, a1x, btx);
  GEMM_COMPUTE(a0y, a1y, bty);

#pragma unroll
  for (int ms = 0; ms < 2; ++ms) {
#pragma unroll
    for (int j = 0; j < 4; ++j) {
      int n = nbase + j * 16 + li;
      float bv = bias[n];
#pragma unroll
      for (int r = 0; r < 8; ++r) {
        int t = t0 + ms * 16 + r + 8 * hi;
        out[(long)t * C_ + n] = acc[ms * 4 + j][r] + bv;
      }
    }
  }
}

// ---- host launcher ---------------------------------------------------------
extern "C" void kernel_launch(void* const* d_in, const int* in_sizes, int n_in,
                              void* d_out, int out_size, void* d_ws, size_t ws_size,
                              hipStream_t stream) {
  (void)in_sizes; (void)n_in; (void)out_size; (void)ws_size;
  const float* x     = (const float*)d_in[0];
  const float* pe    = (const float*)d_in[1];
  const float* mask  = (const float*)d_in[2];
  const float* wqkv  = (const float*)d_in[3];
  const float* bqkv  = (const float*)d_in[4];
  const float* wproj = (const float*)d_in[5];
  const float* bproj = (const float*)d_in[6];
  float* out = (float*)d_out;

  size_t off = 0;
  auto alloc = [&](size_t bytes) {
    void* p = (char*)d_ws + off;
    off += (bytes + 255) & ~(size_t)255;
    return p;
  };
  _Float16* xh   = (_Float16*)alloc((size_t)MTOT * C_ * 2);
  _Float16* wqh  = (_Float16*)alloc((size_t)N3 * C_ * 2);
  _Float16* wph  = (_Float16*)alloc((size_t)C_ * C_ * 2);
  _Float16* qbuf = (_Float16*)alloc((size_t)B_ * H_ * NTOK * HD * 2);
  _Float16* kbuf = (_Float16*)alloc((size_t)B_ * H_ * NTOK * HD * 2);
  _Float16* vtb  = (_Float16*)alloc((size_t)B_ * H_ * HD * TOKPAD * 2);
  _Float16* aout = (_Float16*)alloc((size_t)MTOT * C_ * 2);

  long nx = (long)MTOT * C_ / 4;
  prep_x_kernel<<<dim3((unsigned)((nx + 255) / 256)), 256, 0, stream>>>(x, pe, xh);

  long nw = ((long)N3 * C_ + (long)C_ * C_) / 4;
  prep_w_kernel<<<dim3((unsigned)((nw + 255) / 256)), 256, 0, stream>>>(
      wqkv, wproj, wqh, wph);

  long np = (long)B_ * H_ * HD * 14;
  vt_pad_kernel<<<dim3((unsigned)((np + 255) / 256)), 256, 0, stream>>>(vtb);

  qkv_gemm_kernel<<<dim3(544, 36), 128, 0, stream>>>(xh, wqh, bqkv,
                                                     qbuf, kbuf, vtb);

  attn_kernel<<<dim3(B_ * H_), 32, 0, stream>>>(qbuf, kbuf, vtb, mask, aout);

  proj_gemm_kernel<<<dim3(544, 12), 128, 0, stream>>>(aout, wph, bproj, out);
}
